// PerformerBlock_36670430773880
// MI455X (gfx1250) — matrix operand order repaired and checked by reading, not compiled, not run on hardware
//
#include <hip/hip_runtime.h>
#include <stdint.h>
#include <stddef.h>

// Problem constants (match reference)
#define B_    4
#define S_    4096
#define D_    1024
#define H_    16
#define HD_   64
#define MLP_  4096
#define BS_   (B_ * S_)      // 16384
#define KEPS_ 1e-3f
#define LNEPS_ 1e-6f

typedef __attribute__((ext_vector_type(16))) __bf16 v16bf;
typedef __attribute__((ext_vector_type(8)))  __bf16 v8bf;
typedef __attribute__((ext_vector_type(8)))  float  v8f;
typedef __attribute__((ext_vector_type(4)))  unsigned int v4u;
typedef __attribute__((ext_vector_type(4)))  int v4i;

// ---------- optional CDNA5 async global->LDS path (ASYNCcnt) ----------
#define AS1 __attribute__((address_space(1)))
#define AS3 __attribute__((address_space(3)))
#if defined(__has_builtin)
#if __has_builtin(__builtin_amdgcn_global_load_async_to_lds_b128) && \
    __has_builtin(__builtin_amdgcn_s_wait_asynccnt)
#define HAVE_ASYNC_LDS 1
#endif
#endif
#ifndef HAVE_ASYNC_LDS
#define HAVE_ASYNC_LDS 0
#endif

__device__ __forceinline__ void copy16_g2l(unsigned short* l, const unsigned short* g) {
#if HAVE_ASYNC_LDS
    // global_load_async_to_lds_b128 (ASYNCcnt-tracked DMA, CDNA5 ISA)
    __builtin_amdgcn_global_load_async_to_lds_b128((AS1 v4i*)g, (AS3 v4i*)l, 0, 0);
#else
    *(v4u*)l = *(const v4u*)g;          // global_load_b128 + ds_store_b128
#endif
}
__device__ __forceinline__ void async_join() {
#if HAVE_ASYNC_LDS
    __builtin_amdgcn_s_wait_asynccnt(0);
#endif
}

// ---------- bf16 helpers (bit-level, RNE) ----------
__device__ __forceinline__ unsigned short f2bf(float f) {
    unsigned int x = __float_as_uint(f);
    unsigned int r = x + 0x7fffu + ((x >> 16) & 1u);
    return (unsigned short)(r >> 16);
}
__device__ __forceinline__ float bf2f(unsigned short u) {
    return __uint_as_float(((unsigned int)u) << 16);
}
__device__ __forceinline__ float gelu_f(float x) {
    float x3 = x * x * x;
    return 0.5f * x * (1.0f + tanhf(0.7978845608028654f * (x + 0.044715f * x3)));
}

// ---------- WMMA fragment loaders (bf16, 16x16x32), vectorized ----------
// A (16x32 MxK), row-major in LDS with row stride `ld` (multiple of 8):
// lane l (m=l&15, h=l>>4) needs K = 16h+0..15 contiguous  -> two b128 loads.
__device__ __forceinline__ v16bf load_frag_a(const unsigned short* lds, int row_base,
                                             int ld, int kbase, int lane) {
    int m = lane & 15, half = lane >> 4;
    const unsigned short* p = lds + (row_base + m) * ld + kbase + 16 * half;
    v8bf lo = *(const v8bf*)p;          // K = kbase+16h .. +7
    v8bf hi = *(const v8bf*)(p + 8);    // K = kbase+16h+8 .. +7
    return __builtin_shufflevector(lo, hi, 0, 1, 2, 3, 4, 5, 6, 7,
                                   8, 9, 10, 11, 12, 13, 14, 15);
}
// B (32x16 KxN) sourced from a TRANSPOSED tile Bt[n][k] in LDS:
// lane l (n=l&15, h=l>>4) needs K = kbase+16h+0..15 contiguous -> two b128 loads.
__device__ __forceinline__ v16bf load_frag_bt(const unsigned short* lds, int col_base,
                                              int ld, int kbase, int lane) {
    int n = lane & 15, half = lane >> 4;
    const unsigned short* p = lds + (col_base + n) * ld + kbase + 16 * half;
    v8bf lo = *(const v8bf*)p;
    v8bf hi = *(const v8bf*)(p + 8);
    return __builtin_shufflevector(lo, hi, 0, 1, 2, 3, 4, 5, 6, 7,
                                   8, 9, 10, 11, 12, 13, 14, 15);
}
__device__ __forceinline__ v8f wmma_bf16(v16bf a, v16bf b, v8f c) {
    return __builtin_amdgcn_wmma_f32_16x16x32_bf16(false, a, false, b, (short)0, c,
                                                   false, false);
}

// ---------- LayerNorm (f32 in -> bf16 out) ----------
__global__ __launch_bounds__(256)
void ln_bf16_kernel(const float* __restrict__ x, const float* __restrict__ scale,
                    const float* __restrict__ bias, unsigned short* __restrict__ out) {
    __shared__ float r1[256];
    __shared__ float r2[256];
    const int row = blockIdx.x, tid = threadIdx.x;
    const float* xr = x + (size_t)row * D_;
    float v[4], s1 = 0.f, s2 = 0.f;
#pragma unroll
    for (int i = 0; i < 4; ++i) {
        int c = tid + i * 256;
        v[i] = xr[c];
        s1 += v[i];
        s2 += v[i] * v[i];
    }
    r1[tid] = s1; r2[tid] = s2;
    __syncthreads();
    for (int off = 128; off > 0; off >>= 1) {
        if (tid < off) { r1[tid] += r1[tid + off]; r2[tid] += r2[tid + off]; }
        __syncthreads();
    }
    float mu  = r1[0] * (1.0f / D_);
    float var = r2[0] * (1.0f / D_) - mu * mu;
    float inv = rsqrtf(var + LNEPS_);
#pragma unroll
    for (int i = 0; i < 4; ++i) {
        int c = tid + i * 256;
        out[(size_t)row * D_ + c] = f2bf((v[i] - mu) * inv * scale[c] + bias[c]);
    }
}

// ---------- weight prep: all B operands stored TRANSPOSED [N][K] as bf16 ----------
__global__ void prep_qkv_wT_kernel(const float* __restrict__ Wq, const float* __restrict__ Wk,
                                   const float* __restrict__ Wv, unsigned short* __restrict__ out) {
    size_t i = (size_t)blockIdx.x * 256 + threadIdx.x;     // 3072*1024 outputs
    int c = (int)(i / 1024), d = (int)(i % 1024);          // out[c][d]
    float w = (c < 1024) ? Wq[(size_t)d * 1024 + c]
            : (c < 2048) ? Wk[(size_t)d * 1024 + (c - 1024)]
                         : Wv[(size_t)d * 1024 + (c - 2048)];
    out[i] = f2bf(w);
}
__global__ void prep_qkv_b_kernel(const float* __restrict__ bq, const float* __restrict__ bk,
                                  const float* __restrict__ bv, float* __restrict__ out) {
    int c = blockIdx.x * 256 + threadIdx.x;                // 3072
    out[c] = (c < 1024) ? bq[c] : (c < 2048) ? bk[c - 1024] : bv[c - 2048];
}
// in: [K][N] f32 row-major -> out: [N][K] bf16
__global__ void transpose_bf16_kernel(const float* __restrict__ in,
                                      unsigned short* __restrict__ out, int K, int N) {
    size_t o = (size_t)blockIdx.x * 256 + threadIdx.x;     // over N*K
    int n = (int)(o / K), k = (int)(o % K);
    out[o] = f2bf(in[(size_t)k * N + n]);
}

// ---------- generic bf16 WMMA GEMM: out = act(A[MxK] * Bt[NxK]^T + bias) (+resid) ----------
template <bool GELU_ACT, bool RESID, bool BF16OUT>
__global__ __launch_bounds__(128)
void gemm_bf16_wmma(const unsigned short* __restrict__ A, const unsigned short* __restrict__ Bt,
                    const float* __restrict__ bias, const float* __restrict__ resid,
                    float* __restrict__ outF, unsigned short* __restrict__ outB,
                    int M, int N, int K) {
    __shared__ __align__(16) unsigned short As[64 * 40];
    __shared__ __align__(16) unsigned short Bts[64 * 40];
    const int tid = threadIdx.x, lane = tid & 31, wave = tid >> 5;
    const int row0 = blockIdx.x * 64, col0 = blockIdx.y * 64;

    v8f acc[4];
#pragma unroll
    for (int i = 0; i < 4; ++i)
#pragma unroll
        for (int r = 0; r < 8; ++r) acc[i][r] = 0.0f;

    for (int k0 = 0; k0 < K; k0 += 32) {
        // stage A 64x32 and Bt 64x32, 16B chunks (4 chunks per row)
#pragma unroll
        for (int i = 0; i < 2; ++i) {
            int idx = tid + i * 128;            // 0..255
            int r = idx >> 2, c = idx & 3;
            copy16_g2l(As + r * 40 + 8 * c, A + (size_t)(row0 + r) * K + k0 + 8 * c);
        }
#pragma unroll
        for (int i = 0; i < 2; ++i) {
            int idx = tid + i * 128;
            int r = idx >> 2, c = idx & 3;
            copy16_g2l(Bts + r * 40 + 8 * c, Bt + (size_t)(col0 + r) * K + k0 + 8 * c);
        }
        if (k0 + 32 < K) {                      // global_prefetch_b8 of next K slab
            __builtin_prefetch(A + (size_t)(row0 + (tid >> 1)) * K + k0 + 32, 0, 1);
            __builtin_prefetch(Bt + (size_t)(col0 + (tid >> 1)) * K + k0 + 32, 0, 1);
        }
        async_join();
        __syncthreads();
        v16bf a = load_frag_a(As, wave * 16, 40, 0, lane);
#pragma unroll
        for (int nt = 0; nt < 4; ++nt) {
            v16bf b = load_frag_bt(Bts, nt * 16, 40, 0, lane);
            acc[nt] = wmma_bf16(a, b, acc[nt]);
        }
        __syncthreads();
    }
    // epilogue: C/D layout -> lane l: n = l&15 (+nt*16), VGPR r: m = r + 8*(l>>4)
    const int half = lane >> 4, nn = lane & 15;
#pragma unroll
    for (int nt = 0; nt < 4; ++nt) {
#pragma unroll
        for (int r = 0; r < 8; ++r) {
            int m = r + half * 8;
            int row = row0 + wave * 16 + m;
            int col = col0 + nt * 16 + nn;
            float val = acc[nt][r] + bias[col];
            if (GELU_ACT) val = gelu_f(val);
            if (RESID) val += resid[(size_t)row * N + col];
            if (BF16OUT) outB[(size_t)row * N + col] = f2bf(val);
            else         outF[(size_t)row * N + col] = val;
        }
    }
}

// ---------- phi split: qkv bf16 [BS,3072] -> phi_q [BH,S,64], phi_kT [BH,64,S], vT [BH,64,S] ----------
__global__ void phi_split_kernel(const unsigned short* __restrict__ qkv,
                                 unsigned short* __restrict__ phiq,
                                 unsigned short* __restrict__ phikT,
                                 unsigned short* __restrict__ vT) {
    size_t i = (size_t)blockIdx.x * 256 + threadIdx.x;   // BS*1024
    int bs = (int)(i >> 10), c = (int)(i & 1023);
    int h = c >> 6, m = c & 63;
    int b = bs >> 12, s = bs & (S_ - 1);
    const unsigned short* row = qkv + (size_t)bs * 3072;
    float q = bf2f(row[c]);
    float k = bf2f(row[1024 + c]);
    size_t bh = (size_t)b * H_ + h;
    phiq [(bh * S_ + s) * 64 + m]         = f2bf(fmaxf(q, 0.f) + KEPS_);
    phikT[(bh * 64 + m) * (size_t)S_ + s] = f2bf(fmaxf(k, 0.f) + KEPS_);
    vT   [(bh * 64 + m) * (size_t)S_ + s] = row[2048 + c];
}

// ---------- kv^T = (phi_k^T * V)^T  (per b,h) + ksum ----------
__global__ __launch_bounds__(128)
void kv_ksum_kernel(const unsigned short* __restrict__ phikT,   // [BH,64,S]  (A, row-major)
                    const unsigned short* __restrict__ vT,      // [BH,64,S]  (B^T, row-major)
                    float* __restrict__ kvT,                    // [BH,64(d),64(m)]
                    float* __restrict__ ksum) {                 // [BH,64]
    __shared__ __align__(16) unsigned short As[64 * 40];
    __shared__ __align__(16) unsigned short Bts[64 * 40];
    const int tid = threadIdx.x, lane = tid & 31, wave = tid >> 5;
    const int bh = blockIdx.x;
    const unsigned short* Ab = phikT + (size_t)bh * 64 * S_;
    const unsigned short* Bb = vT + (size_t)bh * 64 * S_;

    v8f acc[4];
#pragma unroll
    for (int i = 0; i < 4; ++i)
#pragma unroll
        for (int r = 0; r < 8; ++r) acc[i][r] = 0.0f;
    float ks = 0.f;

    for (int s0 = 0; s0 < S_; s0 += 32) {
#pragma unroll
        for (int i = 0; i < 2; ++i) {
            int idx = tid + i * 128;
            int r = idx >> 2, c = idx & 3;
            copy16_g2l(As + r * 40 + 8 * c, Ab + (size_t)r * S_ + s0 + 8 * c);
        }
#pragma unroll
        for (int i = 0; i < 2; ++i) {
            int idx = tid + i * 128;
            int r = idx >> 2, c = idx & 3;
            copy16_g2l(Bts + r * 40 + 8 * c, Bb + (size_t)r * S_ + s0 + 8 * c);
        }
        if (s0 + 32 < S_) __builtin_prefetch(Ab + (size_t)(tid >> 1) * S_ + s0 + 32, 0, 1);
        async_join();
        __syncthreads();
        v16bf a = load_frag_a(As, wave * 16, 40, 0, lane);
#pragma unroll
        for (int nt = 0; nt < 4; ++nt) {
            v16bf b = load_frag_bt(Bts, nt * 16, 40, 0, lane);
            acc[nt] = wmma_bf16(a, b, acc[nt]);
        }
        if (tid < 64) {
#pragma unroll
            for (int k = 0; k < 32; ++k) ks += bf2f(As[tid * 40 + k]);
        }
        __syncthreads();
    }
    const int half = lane >> 4, nn = lane & 15;
#pragma unroll
    for (int nt = 0; nt < 4; ++nt)
#pragma unroll
        for (int r = 0; r < 8; ++r) {
            int m = r + half * 8;       // row of kv (feature index)
            int d = nt * 16 + nn;       // col of kv (head dim)
            kvT[((size_t)bh * 64 + d) * 64 + wave * 16 + m] = acc[nt][r];
        }
    if (tid < 64) ksum[(size_t)bh * 64 + tid] = ks;
}

// ---------- attn = (phi_q * kv) / (phi_q . ksum) -> bf16 [BS, H*HD] ----------
__global__ __launch_bounds__(128)
void attn_kernel(const unsigned short* __restrict__ phiq,   // [BH,S,64]
                 const float* __restrict__ kvTf,            // [BH,64(d),64(m)]
                 const float* __restrict__ ksum,            // [BH,64]
                 unsigned short* __restrict__ attn) {       // [BS, 1024]
    __shared__ __align__(16) unsigned short Pq[64 * 72];
    __shared__ __align__(16) unsigned short KVt[64 * 72];
    __shared__ float ksl[64];
    __shared__ float den[64];
    const int tid = threadIdx.x, lane = tid & 31, wave = tid >> 5;
    const int bh = blockIdx.x;
    const int b = bh / H_, h = bh % H_;
    const int row0 = blockIdx.y * 64;
    const unsigned short* Pg = phiq + ((size_t)bh * S_ + row0) * 64;

    // stage phi_q 64x64 (16B chunks, 8 per row)
#pragma unroll
    for (int i = 0; i < 4; ++i) {
        int idx = tid + i * 128;            // 0..511
        int r = idx >> 3, c = idx & 7;
        copy16_g2l(Pq + r * 72 + 8 * c, Pg + (size_t)r * 64 + 8 * c);
    }
    // stage kv^T (f32 -> bf16), already transposed in workspace
    const float* kvb = kvTf + (size_t)bh * 64 * 64;
#pragma unroll
    for (int j = 0; j < 32; ++j) {
        int i = tid + j * 128;              // 0..4095 ; d = i>>6, m = i&63
        KVt[(i >> 6) * 72 + (i & 63)] = f2bf(kvb[i]);
    }
    if (tid < 64) ksl[tid] = ksum[(size_t)bh * 64 + tid];
    async_join();
    __syncthreads();
    if (tid < 64) {
        float d = 0.f;
#pragma unroll
        for (int m = 0; m < 64; ++m) d += bf2f(Pq[tid * 72 + m]) * ksl[m];
        den[tid] = d;
    }
    __syncthreads();

    v8f acc[4];
#pragma unroll
    for (int i = 0; i < 4; ++i)
#pragma unroll
        for (int r = 0; r < 8; ++r) acc[i][r] = 0.0f;
#pragma unroll
    for (int kc = 0; kc < 64; kc += 32) {
        v16bf a = load_frag_a(Pq, wave * 16, 72, kc, lane);
#pragma unroll
        for (int nt = 0; nt < 4; ++nt) {
            v16bf bfrag = load_frag_bt(KVt, nt * 16, 72, kc, lane);
            acc[nt] = wmma_bf16(a, bfrag, acc[nt]);
        }
    }
    const int half = lane >> 4, nn = lane & 15;
#pragma unroll
    for (int nt = 0; nt < 4; ++nt)
#pragma unroll
        for (int r = 0; r < 8; ++r) {
            int m = r + half * 8;
            int s = row0 + wave * 16 + m;
            float val = acc[nt][r] / den[wave * 16 + m];
            attn[((size_t)b * S_ + s) * 1024 + h * 64 + nt * 16 + nn] = f2bf(val);
        }
}

// =====================================================================
extern "C" void kernel_launch(void* const* d_in, const int* in_sizes, int n_in,
                              void* d_out, int out_size, void* d_ws, size_t ws_size,
                              hipStream_t stream) {
    const float* inputs = (const float*)d_in[0];
    const float* ln1_s  = (const float*)d_in[1];
    const float* ln1_b  = (const float*)d_in[2];
    const float* Wq     = (const float*)d_in[3];
    const float* bq     = (const float*)d_in[4];
    const float* Wk     = (const float*)d_in[5];
    const float* bk     = (const float*)d_in[6];
    const float* Wv     = (const float*)d_in[7];
    const float* bv     = (const float*)d_in[8];
    const float* Wo     = (const float*)d_in[9];
    const float* bo     = (const float*)d_in[10];
    const float* ln2_s  = (const float*)d_in[11];
    const float* ln2_b  = (const float*)d_in[12];
    const float* W1     = (const float*)d_in[13];
    const float* b1     = (const float*)d_in[14];
    const float* W2     = (const float*)d_in[15];
    const float* b2     = (const float*)d_in[16];
    float* out          = (float*)d_out;

    unsigned char* ws = (unsigned char*)d_ws;
    size_t off = 0;
    auto take = [&](size_t bytes) -> void* {
        void* p = ws + off;
        off += (bytes + 255) & ~(size_t)255;
        return p;
    };
    unsigned short* x_bf   = (unsigned short*)take((size_t)BS_ * D_ * 2);
    unsigned short* WqkvT  = (unsigned short*)take((size_t)D_ * 3072 * 2);   // [3072][1024]
    float*          bqkv   = (float*)         take(3072 * 4);
    unsigned short* qkv    = (unsigned short*)take((size_t)BS_ * 3072 * 2);
    unsigned short* phiq   = (unsigned short*)take((size_t)BS_ * D_ * 2);
    unsigned short* phikT  = (unsigned short*)take((size_t)BS_ * D_ * 2);
    unsigned short* vTm    = (unsigned short*)take((size_t)BS_ * D_ * 2);
    float*          kvTbuf = (float*)         take((size_t)B_ * H_ * 64 * 64 * 4);
    float*          ksumb  = (float*)         take((size_t)B_ * H_ * 64 * 4);
    unsigned short* attnb  = (unsigned short*)take((size_t)BS_ * D_ * 2);
    unsigned short* WoT    = (unsigned short*)take((size_t)D_ * D_ * 2);     // [1024][1024]
    float*          xres   = (float*)         take((size_t)BS_ * D_ * 4);
    unsigned short* y_bf   = (unsigned short*)take((size_t)BS_ * D_ * 2);
    unsigned short* W1T    = (unsigned short*)take((size_t)D_ * MLP_ * 2);   // [4096][1024]
    unsigned short* W2T    = (unsigned short*)take((size_t)MLP_ * D_ * 2);   // [1024][4096]
    unsigned short* h_bf   = (unsigned short*)take((size_t)BS_ * MLP_ * 2);
    (void)ws_size; (void)n_in; (void)in_sizes; (void)out_size;

    // 1. LN1 -> bf16 x
    ln_bf16_kernel<<<BS_, 256, 0, stream>>>(inputs, ln1_s, ln1_b, x_bf);
    // 2. weight/bias prep (transposed bf16)
    prep_qkv_wT_kernel<<<(D_ * 3072) / 256, 256, 0, stream>>>(Wq, Wk, Wv, WqkvT);
    prep_qkv_b_kernel<<<3072 / 256, 256, 0, stream>>>(bq, bk, bv, bqkv);
    transpose_bf16_kernel<<<(D_ * D_) / 256, 256, 0, stream>>>(Wo, WoT, D_, D_);
    transpose_bf16_kernel<<<(D_ * MLP_) / 256, 256, 0, stream>>>(W1, W1T, D_, MLP_);
    transpose_bf16_kernel<<<(MLP_ * D_) / 256, 256, 0, stream>>>(W2, W2T, MLP_, D_);
    // 3. QKV GEMM: [16384,1024]x[1024,3072] + bias -> bf16 qkv
    gemm_bf16_wmma<false, false, true><<<dim3(BS_ / 64, 3072 / 64), 128, 0, stream>>>(
        x_bf, WqkvT, bqkv, nullptr, nullptr, qkv, BS_, 3072, D_);
    // 4. phi split / relu+eps / head layouts
    phi_split_kernel<<<((size_t)BS_ * D_) / 256, 256, 0, stream>>>(qkv, phiq, phikT, vTm);
    // 5. kv^T + ksum per (b,h)
    kv_ksum_kernel<<<B_ * H_, 128, 0, stream>>>(phikT, vTm, kvTbuf, ksumb);
    // 6. attn = (phi_q kv) / den -> bf16 [BS, 1024]
    attn_kernel<<<dim3(B_ * H_, S_ / 64), 128, 0, stream>>>(phiq, kvTbuf, ksumb, attnb);
    // 7. out-proj + residual -> f32 x_res
    gemm_bf16_wmma<false, true, false><<<dim3(BS_ / 64, D_ / 64), 128, 0, stream>>>(
        attnb, WoT, bo, inputs, xres, nullptr, BS_, D_, D_);
    // 8. LN2 -> bf16 y
    ln_bf16_kernel<<<BS_, 256, 0, stream>>>(xres, ln2_s, ln2_b, y_bf);
    // 9. MLP1 + GELU -> bf16 h
    gemm_bf16_wmma<true, false, true><<<dim3(BS_ / 64, MLP_ / 64), 128, 0, stream>>>(
        y_bf, W1T, b1, nullptr, nullptr, h_bf, BS_, MLP_, D_);
    // 10. MLP2 + residual -> f32 out
    gemm_bf16_wmma<false, true, false><<<dim3(BS_ / 64, D_ / 64), 128, 0, stream>>>(
        h_bf, W2T, b2, xres, out, nullptr, BS_, D_, MLP_);
}
